// FraudAwareAugmentor_core_31413390803481
// MI455X (gfx1250) — matrix operand order repaired and checked by reading, not compiled, not run on hardware
//
#include <hip/hip_runtime.h>
#include <hip/hip_bf16.h>

#define U_DIM 2048
#define I_DIM 16384
#define N_DIM 18432
#define E_DIM 1000000
#define F_DIM 64
#define H_DIM 32
#define KTOP 3
#define IN_DIM 130   // 2F+2

#define TILE 64
#define LDP  72      // padded LDS pitch (bf16 elems): 144B rows -> conflict-free 16B frag loads

typedef __attribute__((ext_vector_type(16))) __bf16 v16bf;
typedef __attribute__((ext_vector_type(8)))  float  v8f;

// ---------------- zero ----------------
__global__ void zero_f4(float4* __restrict__ p, int n4) {
  int i = blockIdx.x * blockDim.x + threadIdx.x;
  int stride = gridDim.x * blockDim.x;
  float4 z = make_float4(0.f, 0.f, 0.f, 0.f);
  for (; i < n4; i += stride) p[i] = z;
}

// ---------------- degrees ----------------
__global__ void deg_kernel(const int* __restrict__ src, const int* __restrict__ dst,
                           float* __restrict__ deg_u, float* __restrict__ deg_i) {
  int e = blockIdx.x * blockDim.x + threadIdx.x;
  if (e >= E_DIM) return;
  atomicAdd(&deg_u[src[e]], 1.0f);
  atomicAdd(&deg_i[dst[e]], 1.0f);
}

__global__ void log1p_kernel(const float* __restrict__ deg_u, float* __restrict__ d_u_out,
                             float* __restrict__ deg_i_inplace) {
  int i = blockIdx.x * blockDim.x + threadIdx.x;
  if (i < U_DIM) d_u_out[i] = log1pf(deg_u[i]);
  if (i < I_DIM) deg_i_inplace[i] = log1pf(deg_i_inplace[i]);
}

// ---------------- per-edge MLP + scatter into A ----------------
__global__ void __launch_bounds__(256) edge_mlp_kernel(
    const float* __restrict__ x, const float* __restrict__ w1,
    const float* __restrict__ b1, const float* __restrict__ w2,
    const float* __restrict__ b2, const int* __restrict__ src,
    const int* __restrict__ dst, const float* __restrict__ d_u,
    const float* __restrict__ d_i, float* __restrict__ A) {
  __shared__ float w1s[H_DIM * IN_DIM];
  __shared__ float w2s[H_DIM];
  __shared__ float b1s[H_DIM];
  __shared__ float b2s;
  for (int t = threadIdx.x; t < H_DIM * IN_DIM; t += blockDim.x) w1s[t] = w1[t];
  if (threadIdx.x < H_DIM) { w2s[threadIdx.x] = w2[threadIdx.x]; b1s[threadIdx.x] = b1[threadIdx.x]; }
  if (threadIdx.x == 0) b2s = b2[0];
  __syncthreads();

  int e = blockIdx.x * blockDim.x + threadIdx.x;
  if (e >= E_DIM) return;
  int u = src[e], it = dst[e];
  const float* xu = x + (size_t)u * F_DIM;
  const float* xi = x + (size_t)(U_DIM + it) * F_DIM;
  float du = d_u[u], di = d_i[it];

  float acc[H_DIM];
  #pragma unroll
  for (int j = 0; j < H_DIM; ++j)
    acc[j] = b1s[j] + du * w1s[j * IN_DIM + 64] + di * w1s[j * IN_DIM + 129];

  for (int f = 0; f < F_DIM; ++f) {
    float xv = xu[f], iv = xi[f];
    #pragma unroll
    for (int j = 0; j < H_DIM; ++j)
      acc[j] = fmaf(xv, w1s[j * IN_DIM + f], fmaf(iv, w1s[j * IN_DIM + 65 + f], acc[j]));
  }

  float s = b2s;
  #pragma unroll
  for (int j = 0; j < H_DIM; ++j) s += fmaxf(acc[j], 0.f) * w2s[j];
  float w = 1.f / (1.f + __expf(-s));
  atomicAdd(&A[(size_t)u * I_DIM + it], w);
}

// ---------------- C = Dh (A A^T) Dh via split-bf16 WMMA ----------------
__global__ void __launch_bounds__(256) gemm_aat_wmma(
    const float* __restrict__ A, const float* __restrict__ d_u,
    float* __restrict__ C) {
  __shared__ __align__(16) __bf16 rhi[TILE * LDP];
  __shared__ __align__(16) __bf16 rlo[TILE * LDP];
  __shared__ __align__(16) __bf16 chi[TILE * LDP];
  __shared__ __align__(16) __bf16 clo[TILE * LDP];

  const int tid  = threadIdx.x;
  const int lane = tid & 31;
  const int wid  = tid >> 5;
  const int rb = blockIdx.y, cb = blockIdx.x;
  const int mi  = wid & 3;              // 16-row subtile of the 64x64 C tile
  const int nj0 = (wid >> 2) * 2;       // two 16-col subtiles per wave
  const int nj1 = nj0 + 1;

  v8f acc0 = {}, acc1 = {};

  for (int kc = 0; kc < I_DIM; kc += TILE) {
    // cooperative load of row-block and col-block tiles; split fp32 -> bf16 hi/lo
    #pragma unroll
    for (int itr = 0; itr < 4; ++itr) {
      int fi  = tid + 256 * itr;        // float4 index within 64x64 tile
      int row = fi >> 4;
      int cg  = fi & 15;
      const float4 a4 = *(const float4*)&A[((size_t)(rb * TILE + row)) * I_DIM + kc + cg * 4];
      const float4 b4 = *(const float4*)&A[((size_t)(cb * TILE + row)) * I_DIM + kc + cg * 4];
      if (kc + TILE < I_DIM) {
        __builtin_prefetch(&A[((size_t)(rb * TILE + row)) * I_DIM + kc + TILE + cg * 4], 0, 1);
        __builtin_prefetch(&A[((size_t)(cb * TILE + row)) * I_DIM + kc + TILE + cg * 4], 0, 1);
      }
      #pragma unroll
      for (int q = 0; q < 4; ++q) {
        float av = ((const float*)&a4)[q];
        __bf16 ah = (__bf16)av;
        rhi[row * LDP + cg * 4 + q] = ah;
        rlo[row * LDP + cg * 4 + q] = (__bf16)(av - (float)ah);
        float bv = ((const float*)&b4)[q];
        __bf16 bh = (__bf16)bv;
        chi[row * LDP + cg * 4 + q] = bh;
        clo[row * LDP + cg * 4 + q] = (__bf16)(bv - (float)bh);
      }
    }
    __syncthreads();

    #pragma unroll
    for (int ks = 0; ks < 2; ++ks) {      // two K=32 WMMA steps per 64-wide chunk
      const int k0 = ks * 32;
      // A fragment (16x32 bf16, MxK): lane<16 holds K 0..15, lane>=16 K 16..31
      const int am    = lane & 15;
      const int abase = k0 + ((lane & 16) ? 16 : 0);
      const int arow  = (mi * 16 + am) * LDP;
      v16bf a_hi, a_lo, b0_hi, b0_lo, b1_hi, b1_lo;
      *((uint4*)&a_hi)       = *(const uint4*)&rhi[arow + abase];
      *(((uint4*)&a_hi) + 1) = *(const uint4*)&rhi[arow + abase + 8];
      *((uint4*)&a_lo)       = *(const uint4*)&rlo[arow + abase];
      *(((uint4*)&a_lo) + 1) = *(const uint4*)&rlo[arow + abase + 8];
      // B fragment (32x16 bf16, KxN): B[k][n] = A_coltile[n][k]; lane<16 -> K 0..15, lane>=16 -> K 16..31
      const int bn    = lane & 15;
      const int bbase = k0 + ((lane & 16) ? 16 : 0);
      const int brow0 = (nj0 * 16 + bn) * LDP;
      const int brow1 = (nj1 * 16 + bn) * LDP;
      *((uint4*)&b0_hi)       = *(const uint4*)&chi[brow0 + bbase];
      *(((uint4*)&b0_hi) + 1) = *(const uint4*)&chi[brow0 + bbase + 8];
      *((uint4*)&b0_lo)       = *(const uint4*)&clo[brow0 + bbase];
      *(((uint4*)&b0_lo) + 1) = *(const uint4*)&clo[brow0 + bbase + 8];
      *((uint4*)&b1_hi)       = *(const uint4*)&chi[brow1 + bbase];
      *(((uint4*)&b1_hi) + 1) = *(const uint4*)&chi[brow1 + bbase + 8];
      *((uint4*)&b1_lo)       = *(const uint4*)&clo[brow1 + bbase];
      *(((uint4*)&b1_lo) + 1) = *(const uint4*)&clo[brow1 + bbase + 8];

      // split-bf16 product: hi*hi + hi*lo + lo*hi  (~fp32 accuracy)
      acc0 = __builtin_amdgcn_wmma_f32_16x16x32_bf16(false, a_hi, false, b0_hi, (short)0, acc0, false, false);
      acc0 = __builtin_amdgcn_wmma_f32_16x16x32_bf16(false, a_hi, false, b0_lo, (short)0, acc0, false, false);
      acc0 = __builtin_amdgcn_wmma_f32_16x16x32_bf16(false, a_lo, false, b0_hi, (short)0, acc0, false, false);
      acc1 = __builtin_amdgcn_wmma_f32_16x16x32_bf16(false, a_hi, false, b1_hi, (short)0, acc1, false, false);
      acc1 = __builtin_amdgcn_wmma_f32_16x16x32_bf16(false, a_hi, false, b1_lo, (short)0, acc1, false, false);
      acc1 = __builtin_amdgcn_wmma_f32_16x16x32_bf16(false, a_lo, false, b1_hi, (short)0, acc1, false, false);
    }
    __syncthreads();
  }

  // epilogue: C/D layout f32 16x16 -> lane holds col N=lane%16, rows M=r+(lane>=16?8:0)
  const int n    = lane & 15;
  const int moff = (lane & 16) ? 8 : 0;
  #pragma unroll
  for (int r = 0; r < 8; ++r) {
    int grow = rb * TILE + mi * 16 + r + moff;
    float dr = sqrtf(d_u[grow]);
    int gc0 = cb * TILE + nj0 * 16 + n;
    int gc1 = cb * TILE + nj1 * 16 + n;
    C[(size_t)grow * U_DIM + gc0] = acc0[r] * dr * sqrtf(d_u[gc0]);
    C[(size_t)grow * U_DIM + gc1] = acc1[r] * dr * sqrtf(d_u[gc1]);
  }
}

// ---------------- per-row top-3 + symmetric scatter ----------------
__global__ void __launch_bounds__(256) topk_scatter(const float* __restrict__ C,
                                                    float* __restrict__ S) {
  int wid = threadIdx.x >> 5, lane = threadIdx.x & 31;
  int r = blockIdx.x * 8 + wid;
  if (r >= U_DIM) return;

  float v0 = -1e30f, v1 = -1e30f, v2 = -1e30f;
  int   i0 = 1 << 30, i1 = 1 << 30, i2 = 1 << 30;
  for (int c = lane; c < U_DIM; c += 32) {
    float v = C[(size_t)r * U_DIM + c];
    if (v > 0.f) {
      if (v > v0 || (v == v0 && c < i0))      { v2 = v1; i2 = i1; v1 = v0; i1 = i0; v0 = v; i0 = c; }
      else if (v > v1 || (v == v1 && c < i1)) { v2 = v1; i2 = i1; v1 = v;  i1 = c; }
      else if (v > v2 || (v == v2 && c < i2)) { v2 = v;  i2 = c; }
    }
  }
  for (int off = 16; off > 0; off >>= 1) {
    float ov[3] = { __shfl_xor(v0, off, 32), __shfl_xor(v1, off, 32), __shfl_xor(v2, off, 32) };
    int   oi[3] = { __shfl_xor(i0, off, 32), __shfl_xor(i1, off, 32), __shfl_xor(i2, off, 32) };
    #pragma unroll
    for (int t = 0; t < 3; ++t) {
      float v = ov[t]; int c = oi[t];
      if (v > v0 || (v == v0 && c < i0))      { v2 = v1; i2 = i1; v1 = v0; i1 = i0; v0 = v; i0 = c; }
      else if (v > v1 || (v == v1 && c < i1)) { v2 = v1; i2 = i1; v1 = v;  i1 = c; }
      else if (v > v2 || (v == v2 && c < i2)) { v2 = v;  i2 = c; }
    }
  }
  if (lane == 0) {
    float vv[3] = { v0, v1, v2 };
    int   ii[3] = { i0, i1, i2 };
    #pragma unroll
    for (int t = 0; t < KTOP; ++t) {
      if (vv[t] > 0.f) {
        float h = 0.5f * vv[t];
        atomicAdd(&S[(size_t)r * U_DIM + ii[t]], h);   // T[r, idx] += vhalf
        atomicAdd(&S[(size_t)ii[t] * U_DIM + r], h);   // + transpose => S = T + T^T
      }
    }
  }
}

extern "C" void kernel_launch(void* const* d_in, const int* in_sizes, int n_in,
                              void* d_out, int out_size, void* d_ws, size_t ws_size,
                              hipStream_t stream) {
  const float* x   = (const float*)d_in[0];
  const float* w1  = (const float*)d_in[1];
  const float* b1  = (const float*)d_in[2];
  const float* w2  = (const float*)d_in[3];
  const float* b2  = (const float*)d_in[4];
  const int* esrc  = (const int*)d_in[5];
  const int* edst  = (const int*)d_in[6];
  // d_in[7] = user_idx (identity arange), d_in[8] = num_nodes : unused

  float* out = (float*)d_out;
  float* S   = out;                                  // [U, U]
  float* d_u = out + (size_t)U_DIM * U_DIM;          // [U] (second output)

  float* ws    = (float*)d_ws;
  float* A     = ws;                                 // [U, I] fp32
  float* C     = A + (size_t)U_DIM * I_DIM;          // [U, U] fp32
  float* deg_u = C + (size_t)U_DIM * U_DIM;          // [U] raw counts
  float* deg_i = deg_u + U_DIM;                      // [I] raw counts -> log1p in place

  size_t nz = (size_t)U_DIM * I_DIM + (size_t)U_DIM * U_DIM + U_DIM + I_DIM;
  zero_f4<<<2048, 256, 0, stream>>>((float4*)ws, (int)(nz / 4));
  zero_f4<<<1024, 256, 0, stream>>>((float4*)S, (int)((size_t)U_DIM * U_DIM / 4));

  deg_kernel<<<(E_DIM + 255) / 256, 256, 0, stream>>>(esrc, edst, deg_u, deg_i);
  log1p_kernel<<<(I_DIM + 255) / 256, 256, 0, stream>>>(deg_u, d_u, deg_i);
  edge_mlp_kernel<<<(E_DIM + 255) / 256, 256, 0, stream>>>(x, w1, b1, w2, b2, esrc, edst, d_u, deg_i, A);

  dim3 g(U_DIM / TILE, U_DIM / TILE);
  gemm_aat_wmma<<<g, 256, 0, stream>>>(A, d_u, C);
  topk_scatter<<<U_DIM / 8, 256, 0, stream>>>(C, S);
}
